// SparseBKCore_48421461295383
// MI455X (gfx1250) — matrix operand order repaired and checked by reading, not compiled, not run on hardware
//
#include <hip/hip_runtime.h>

// ---------------------------------------------------------------------------
// Types for CDNA5 WMMA (wave32, 16x16x32 bf16 -> f32)
// ---------------------------------------------------------------------------
typedef __attribute__((ext_vector_type(16))) __bf16          v16bf;
typedef __attribute__((ext_vector_type(16))) unsigned short  v16u16;
typedef __attribute__((ext_vector_type(8)))  unsigned short  v8u16;
typedef __attribute__((ext_vector_type(8)))  float           v8f;

// pointee type expected by the async-LDS builtin: int __vector(4)
typedef int v4i_vs __attribute__((vector_size(16)));
typedef __attribute__((address_space(1))) v4i_vs gas_v4i;   // global
typedef __attribute__((address_space(3))) v4i_vs las_v4i;   // LDS

__device__ __forceinline__ unsigned short f2bf(float f) {
    unsigned int u = __builtin_bit_cast(unsigned int, f);
    unsigned int r = u + 0x7FFFu + ((u >> 16) & 1u);   // round-to-nearest-even
    return (unsigned short)(r >> 16);
}
__device__ __forceinline__ float bf2f(unsigned short h) {
    return __builtin_bit_cast(float, ((unsigned int)h) << 16);
}
__device__ __forceinline__ v8f wmma_bf16(v16bf a, v16bf b, v8f c) {
    return __builtin_amdgcn_wmma_f32_16x16x32_bf16(false, a, false, b, (short)0, c, false, false);
}

// ---- CDNA5 async global->LDS copy (16B per lane), ASYNCcnt-tracked ----------
#if __has_builtin(__builtin_amdgcn_global_load_async_to_lds_b128)
__device__ __forceinline__ void async_cp16(const void* g, void* l) {
    __builtin_amdgcn_global_load_async_to_lds_b128(
        (gas_v4i*)g, (las_v4i*)l, 0, 0);
}
#else
__device__ __forceinline__ void async_cp16(const void* g, void* l) {
    unsigned lds_off = (unsigned)(unsigned long long)
        (__attribute__((address_space(3))) void*)l;
    unsigned long long ga = (unsigned long long)g;
    asm volatile("global_load_async_to_lds_b128 %0, %1, off"
                 :: "v"(lds_off), "v"(ga) : "memory");
}
#endif

#if __has_builtin(__builtin_amdgcn_s_wait_asynccnt)
#define WAIT_ASYNC(n) __builtin_amdgcn_s_wait_asynccnt(n)
#else
#define WAIT_ASYNC(n) asm volatile("s_wait_asynccnt %0" :: "i"(n) : "memory")
#endif

#define BB 32
#define NN 2048
#define DD 1024
#define HH 512

// ---------------------------------------------------------------------------
// Kernel 0: zero score accumulator + mask-sum scalar in workspace
// ---------------------------------------------------------------------------
__global__ void k_init(float* __restrict__ scoresBuf, float* __restrict__ sumMask) {
    int i = blockIdx.x * 256 + threadIdx.x;
    if (i < BB * NN) scoresBuf[i] = 0.f;
    if (i == BB * NN) *sumMask = 0.f;
}

// ---------------------------------------------------------------------------
// Kernel 1: W1 (D,H) f32 -> W1T (H,D) split into bf16 hi/lo  (2 MB, L2-resident)
// ---------------------------------------------------------------------------
__global__ __launch_bounds__(256) void k_prep(const float* __restrict__ W1,
                                              unsigned short* __restrict__ w1th,
                                              unsigned short* __restrict__ w1tl) {
    int tid = blockIdx.x * 256 + threadIdx.x;    // < H*D
    int h = tid >> 10, d = tid & 1023;
    float w = W1[d * HH + h];
    unsigned short hi = f2bf(w);
    w1th[tid] = hi;
    w1tl[tid] = f2bf(w - bf2f(hi));
}

// ---------------------------------------------------------------------------
// Kernel 2: scores GEMM via split-bf16 WMMA with double-buffered async staging.
// Block = 256 thr (8 waves): 64 rows x 256 hidden cols.  Grid = 1024 row-blocks x 2 col-halves.
// Per 32-wide K-chunk: x (f32) and W1T hi/lo (bf16) land in LDS via
// GLOBAL_LOAD_ASYNC_TO_LDS_B128; next chunk is prefetched while WMMAs run.
// ---------------------------------------------------------------------------
__global__ __launch_bounds__(256) void k_gemm(const float* __restrict__ x,
                                              const unsigned short* __restrict__ w1th,
                                              const unsigned short* __restrict__ w1tl,
                                              const float* __restrict__ b1,
                                              const float* __restrict__ W2,
                                              float* __restrict__ scoresBuf) {
    __shared__ __align__(16) float          xf[2][64][32];
    __shared__ __align__(16) unsigned short wh[2][256][32], wl[2][256][32];
    __shared__ float sScores[64];

    const int t    = threadIdx.x;
    const int lane = t & 31;
    const int wv   = t >> 5;
    const int wm   = wv >> 1;            // 0..3  (M tile)
    const int wq   = wv & 1;             // 0..1  (128-col quarter)
    const int rowBase = (blockIdx.x >> 1) * 64;
    const int colBase = (blockIdx.x & 1) * 256;
    const int half = lane >> 4;
    const int l16  = lane & 15;
    const int mRow = wm * 16 + l16;

    if (t < 64) sScores[t] = 0.f;

    // per-thread staging slices (10 async b128 per thread per chunk)
    const int xr = t >> 2, xc = (t & 3) * 8;
    auto issue = [&](int buf, int kk2) {
        const float* xs = x + (size_t)(rowBase + xr) * DD + kk2 + xc;
        async_cp16(xs,     &xf[buf][xr][xc]);
        async_cp16(xs + 4, &xf[buf][xr][xc + 4]);
        const unsigned short* sh = w1th + (size_t)(colBase + t) * DD + kk2;
        const unsigned short* sl = w1tl + (size_t)(colBase + t) * DD + kk2;
#pragma unroll
        for (int q = 0; q < 4; q++) {
            async_cp16(sh + q * 8, &wh[buf][t][q * 8]);
            async_cp16(sl + q * 8, &wl[buf][t][q * 8]);
        }
    };

    v8f acc[8];
    const v8f vz = {0.f, 0.f, 0.f, 0.f, 0.f, 0.f, 0.f, 0.f};
#pragma unroll
    for (int i = 0; i < 8; i++) acc[i] = vz;

    issue(0, 0);
    for (int kk = 0; kk < DD; kk += 32) {
        const int cur = (kk >> 5) & 1;
        if (kk + 32 < DD) {
            issue(cur ^ 1, kk + 32);   // prefetch next chunk
            WAIT_ASYNC(10);            // drain current chunk's 10 copies (in-order)
        } else {
            WAIT_ASYNC(0);
        }
        __syncthreads();

        // ---- A operand per ISA 16-bit A(16x32) layout: K(e) = (e>=8?16:0)+8*half+(e&7)
        float4 a0 = *(const float4*)&xf[cur][mRow][8 * half];
        float4 a1 = *(const float4*)&xf[cur][mRow][8 * half + 4];
        float4 a2 = *(const float4*)&xf[cur][mRow][16 + 8 * half];
        float4 a3 = *(const float4*)&xf[cur][mRow][16 + 8 * half + 4];
        float af[16] = {a0.x, a0.y, a0.z, a0.w, a1.x, a1.y, a1.z, a1.w,
                        a2.x, a2.y, a2.z, a2.w, a3.x, a3.y, a3.z, a3.w};
        v16u16 au, alu;
#pragma unroll
        for (int e = 0; e < 16; e++) {
            unsigned short hi = f2bf(af[e]);
            au[e]  = hi;
            alu[e] = f2bf(af[e] - bf2f(hi));
        }
        v16bf ah = __builtin_bit_cast(v16bf, au);
        v16bf al = __builtin_bit_cast(v16bf, alu);

#pragma unroll
        for (int tt = 0; tt < 8; tt++) {
            int nc = wq * 128 + tt * 16 + l16;   // local col in staged chunk
            // ---- B operand per ISA 16-bit B(32x16) layout: K(e) = (e>=8?16:0)+8*half+(e&7)
            v16u16 bu, blu;
            {
                v8u16 p0 = *(const v8u16*)&wh[cur][nc][8 * half];
                v8u16 p1 = *(const v8u16*)&wh[cur][nc][16 + 8 * half];
                v8u16 q0 = *(const v8u16*)&wl[cur][nc][8 * half];
                v8u16 q1 = *(const v8u16*)&wl[cur][nc][16 + 8 * half];
#pragma unroll
                for (int e = 0; e < 8; e++) { bu[e] = p0[e]; bu[e + 8] = p1[e]; blu[e] = q0[e]; blu[e + 8] = q1[e]; }
            }
            v16bf bh = __builtin_bit_cast(v16bf, bu);
            v16bf bl = __builtin_bit_cast(v16bf, blu);
            // split-bf16 3-term product ~ fp32 accuracy
            acc[tt] = wmma_bf16(ah, bh, acc[tt]);
            acc[tt] = wmma_bf16(ah, bl, acc[tt]);
            acc[tt] = wmma_bf16(al, bh, acc[tt]);
        }
        __syncthreads();   // all waves done reading `cur` before it is re-issued
    }

    // ---- epilogue: hidden = acc + b1, relu, dot with W2; reduce over N
    float sacc[8] = {0.f, 0.f, 0.f, 0.f, 0.f, 0.f, 0.f, 0.f};
#pragma unroll
    for (int tt = 0; tt < 8; tt++) {
        int n = colBase + wq * 128 + tt * 16 + l16;
        float b1v = b1[n], w2v = W2[n];
#pragma unroll
        for (int r = 0; r < 8; r++) {
            float h = fmaxf(acc[tt][r] + b1v, 0.f);
            sacc[r] = fmaf(h, w2v, sacc[r]);
        }
    }
#pragma unroll
    for (int d = 1; d < 16; d <<= 1)
#pragma unroll
        for (int r = 0; r < 8; r++) sacc[r] += __shfl_xor(sacc[r], d);
    if (l16 == 0) {
#pragma unroll
        for (int r = 0; r < 8; r++) {
            int m = wm * 16 + r + 8 * half;     // C/D layout: M = r + 8*(lane/16)
            atomicAdd(&sScores[m], sacc[r]);
        }
    }
    __syncthreads();
    if (t < 64) atomicAdd(&scoresBuf[rowBase + t], sScores[t]);
}

// ---------------------------------------------------------------------------
// Kernel 3: gumbel sigmoid mask + block-reduced mask sum
// ---------------------------------------------------------------------------
__global__ __launch_bounds__(256) void k_mask(const float* __restrict__ scoresBuf,
                                              const float* __restrict__ b2,
                                              const float* __restrict__ gum,
                                              float* __restrict__ maskOut,
                                              float* __restrict__ maskWs,
                                              float* __restrict__ sumMask) {
    __shared__ float bsum;
    int t = threadIdx.x;
    if (t == 0) bsum = 0.f;
    __syncthreads();
    int r = blockIdx.x * 256 + t;
    float score = scoresBuf[r] + b2[0];
    float u = gum[r];
    float g = -logf(-logf(u + 1e-10f) + 1e-10f);
    float soft = 1.f / (1.f + expf(-(score + g)));     // TAU = 1
    float hard = soft > 0.5f ? 1.f : 0.f;
    float m = (hard - soft) + soft;                    // straight-through forward value
    maskOut[r] = m;
    maskWs[r] = m;
    float s = m;
#pragma unroll
    for (int d = 1; d < 32; d <<= 1) s += __shfl_xor(s, d);
    if ((t & 31) == 0) atomicAdd(&bsum, s);
    __syncthreads();
    if (t == 0) atomicAdd(sumMask, bsum);
}

// ---------------------------------------------------------------------------
// Kernel 4: skip[i] = any_b(mask[b,i] < 0.5); also finalize sparsity scalar
// ---------------------------------------------------------------------------
__global__ void k_skip(const float* __restrict__ maskWs, int* __restrict__ skip,
                       const float* __restrict__ sumMask, float* __restrict__ outSparsity) {
    int i = blockIdx.x * 256 + threadIdx.x;
    if (i < NN) {
        int any = 0;
        for (int b = 0; b < BB; b++) any |= (maskWs[b * NN + i] < 0.5f) ? 1 : 0;
        skip[i] = any;
    }
    if (i == 0) *outSparsity = 1.f - (*sumMask) * (1.f / (float)(BB * NN));
}

// ---------------------------------------------------------------------------
// Kernel 5: _bk_sparse — complex128 linear recurrences via 2x2 matrix wave-scan.
// One wave (32 lanes) per batch; each lane folds 64 steps, shfl_up matrix scan,
// then replays to emit theta (forward) and phi*theta/det -> feats (backward).
// ---------------------------------------------------------------------------
struct cd { double re, im; };
__device__ __forceinline__ cd cmul(cd a, cd b) { cd r; r.re = a.re * b.re - a.im * b.im; r.im = a.re * b.im + a.im * b.re; return r; }
__device__ __forceinline__ cd cadd(cd a, cd b) { cd r; r.re = a.re + b.re; r.im = a.im + b.im; return r; }
__device__ __forceinline__ cd csub(cd a, cd b) { cd r; r.re = a.re - b.re; r.im = a.im - b.im; return r; }
__device__ __forceinline__ cd cscale(double s, cd a) { cd r; r.re = s * a.re; r.im = s * a.im; return r; }

struct mat2 { cd m00, m01, m10, m11; };
__device__ __forceinline__ mat2 mident() {
    mat2 r; r.m00 = {1.0, 0.0}; r.m01 = {0.0, 0.0}; r.m10 = {0.0, 0.0}; r.m11 = {1.0, 0.0}; return r;
}
__device__ __forceinline__ mat2 mmul(const mat2& A, const mat2& B) {
    mat2 r;
    r.m00 = cadd(cmul(A.m00, B.m00), cmul(A.m01, B.m10));
    r.m01 = cadd(cmul(A.m00, B.m01), cmul(A.m01, B.m11));
    r.m10 = cadd(cmul(A.m10, B.m00), cmul(A.m11, B.m10));
    r.m11 = cadd(cmul(A.m10, B.m01), cmul(A.m11, B.m11));
    return r;
}
__device__ __forceinline__ cd shfl_up_cd(cd c, int d) {
    cd r; r.re = __shfl_up(c.re, d); r.im = __shfl_up(c.im, d); return r;
}
__device__ __forceinline__ mat2 shfl_up_mat(const mat2& m, int d) {
    mat2 r; r.m00 = shfl_up_cd(m.m00, d); r.m01 = shfl_up_cd(m.m01, d);
    r.m10 = shfl_up_cd(m.m10, d); r.m11 = shfl_up_cd(m.m11, d); return r;
}
// apply step matrix M = [[a,-c],[1,0]] on the left of P
__device__ __forceinline__ void applyM(cd a, double c, mat2& P) {
    cd t0 = csub(cmul(a, P.m00), cscale(c, P.m10));
    cd t1 = csub(cmul(a, P.m01), cscale(c, P.m11));
    P.m10 = P.m00; P.m11 = P.m01; P.m00 = t0; P.m01 = t1;
}

__global__ __launch_bounds__(32) void k_scan(const float* __restrict__ v,
                                             const int* __restrict__ skip,
                                             double* __restrict__ theta,
                                             float* __restrict__ feats) {
    const int b = blockIdx.x;
    const int lane = threadIdx.x;
    const float* vb = v + (size_t)b * NN;
    double* th = theta + (size_t)b * (NN + 1) * 2;

    // ================= forward: theta =================
    mat2 P = mident();
    const int k0 = lane * 64;
    for (int s = 0; s < 64; s++) {
        int k = k0 + s; if (k >= NN - 1) break;
        int i = k + 1;
        cd a = {(double)vb[i] - 2.0, -1.0};
        double c = skip[i] ? 0.0 : 1.0;
        applyM(a, c, P);
    }
    mat2 Acc = P;
#pragma unroll
    for (int d = 1; d < 32; d <<= 1) {
        mat2 Q = shfl_up_mat(Acc, d);
        if (lane >= d) Acc = mmul(Acc, Q);
    }
    mat2 E = shfl_up_mat(Acc, 1);
    if (lane == 0) E = mident();

    cd s1 = {(double)vb[0] - 2.0, -1.0};   // theta_1 = ash[0]
    cd s0 = {1.0, 0.0};                    // theta_0 = 1
    cd u = cadd(cmul(E.m00, s1), cmul(E.m01, s0));   // theta_{k0+1}
    cd w = cadd(cmul(E.m10, s1), cmul(E.m11, s0));   // theta_{k0}
    if (lane == 0) { th[0] = 1.0; th[1] = 0.0; th[2] = s1.re; th[3] = s1.im; }
    for (int s = 0; s < 64; s++) {
        int k = k0 + s; if (k >= NN - 1) break;
        int i = k + 1;
        cd a = {(double)vb[i] - 2.0, -1.0};
        double c = skip[i] ? 0.0 : 1.0;
        cd tn = csub(cmul(a, u), cscale(c, w));
        th[(size_t)(k + 2) * 2] = tn.re;
        th[(size_t)(k + 2) * 2 + 1] = tn.im;
        w = u; u = tn;
    }
    cd det; det.re = __shfl(u.re, 31); det.im = __shfl(u.im, 31);   // theta_N
    double mag = sqrt(det.re * det.re + det.im * det.im);
    if (mag < 1e-9) {
        double m2 = fmax(mag, 1e-30);
        det.re = det.re / m2 * 1e-9;
        det.im = det.im / m2 * 1e-9;
    }
    cd den = {det.re + 1e-18, det.im};
    double nn2 = den.re * den.re + den.im * den.im;
    cd inv = {den.re / nn2, -den.im / nn2};
    __threadfence();   // theta stores visible before cross-lane reads below

    // ================= backward: phi, G, feats =================
    P = mident();
    const int i0 = (NN - 1) - lane * 64;
    for (int s = 0; s < 64; s++) {
        int i = i0 - s;
        cd a = {(double)vb[i] - 2.0, -1.0};
        double c = (skip[i] || i == NN - 1) ? 0.0 : 1.0;
        applyM(a, c, P);
    }
    Acc = P;
#pragma unroll
    for (int d = 1; d < 32; d <<= 1) {
        mat2 Q = shfl_up_mat(Acc, d);
        if (lane >= d) Acc = mmul(Acc, Q);
    }
    E = shfl_up_mat(Acc, 1);
    if (lane == 0) E = mident();
    cd p1 = E.m00;   // (p_{i0+1}, p_{i0+2}) = E * (1,0)^T
    cd p2 = E.m10;
    for (int s = 0; s < 64; s++) {
        int i = i0 - s;
        cd a = {(double)vb[i] - 2.0, -1.0};
        double c = (skip[i] || i == NN - 1) ? 0.0 : 1.0;
        cd p = csub(cmul(a, p1), cscale(c, p2));
        cd tth = {th[(size_t)i * 2], th[(size_t)i * 2 + 1]};
        cd G = cmul(cmul(tth, p), inv);
        float* f = feats + ((size_t)b * NN + i) * 2;
        f[0] = (float)G.re;
        f[1] = (float)G.im;
        p2 = p1; p1 = p;
    }
}

// ---------------------------------------------------------------------------
// Kernel 6: conv1d stack (2->16->16->2, k=3, pad=1) with per-point recompute,
// then final = where(mask>0.5, feats, interp)
// ---------------------------------------------------------------------------
__global__ __launch_bounds__(256) void k_conv(const float* __restrict__ feats,
                                              const float* __restrict__ maskWs,
                                              const float* __restrict__ cw1, const float* __restrict__ cb1,
                                              const float* __restrict__ cw2, const float* __restrict__ cb2,
                                              const float* __restrict__ cw3, const float* __restrict__ cb3,
                                              float* __restrict__ outFinal) {
    __shared__ float w1s[96], b1s[16], w2s[768], b2s[16], w3s[96], b3s[2];
    int t = threadIdx.x;
    for (int j = t; j < 96; j += 256)  w1s[j] = cw1[j];
    for (int j = t; j < 16; j += 256)  b1s[j] = cb1[j];
    for (int j = t; j < 768; j += 256) w2s[j] = cw2[j];
    for (int j = t; j < 16; j += 256)  b2s[j] = cb2[j];
    for (int j = t; j < 96; j += 256)  w3s[j] = cw3[j];
    for (int j = t; j < 2; j += 256)   b3s[j] = cb3[j];
    __syncthreads();

    int idx = blockIdx.x * 256 + t;
    int b = idx >> 11, n = idx & (NN - 1);

    float fm[7][2];
#pragma unroll
    for (int q = 0; q < 7; q++) {
        int pos = n + q - 3;
        if (pos >= 0 && pos < NN) {
            float m = maskWs[b * NN + pos];
            const float* fp = feats + ((size_t)b * NN + pos) * 2;
            fm[q][0] = fp[0] * m;
            fm[q][1] = fp[1] * m;
        } else { fm[q][0] = 0.f; fm[q][1] = 0.f; }
    }
    float h1[5][16];
#pragma unroll
    for (int p = 0; p < 5; p++)
#pragma unroll
        for (int oc = 0; oc < 16; oc++) {
            float a = b1s[oc];
#pragma unroll
            for (int ic = 0; ic < 2; ic++)
#pragma unroll
                for (int q = 0; q < 3; q++) a = fmaf(w1s[oc * 6 + ic * 3 + q], fm[p + q][ic], a);
            h1[p][oc] = fmaxf(a, 0.f);
        }
    float h2[3][16];
#pragma unroll
    for (int p = 0; p < 3; p++)
#pragma unroll
        for (int oc = 0; oc < 16; oc++) {
            float a = b2s[oc];
#pragma unroll
            for (int ic = 0; ic < 16; ic++)
#pragma unroll
                for (int q = 0; q < 3; q++) a = fmaf(w2s[oc * 48 + ic * 3 + q], h1[p + q][ic], a);
            h2[p][oc] = fmaxf(a, 0.f);
        }
    float m0 = maskWs[b * NN + n];
#pragma unroll
    for (int ch = 0; ch < 2; ch++) {
        float a = b3s[ch];
#pragma unroll
        for (int ic = 0; ic < 16; ic++)
#pragma unroll
            for (int q = 0; q < 3; q++) a = fmaf(w3s[ch * 48 + ic * 3 + q], h2[q][ic], a);
        float f = feats[((size_t)b * NN + n) * 2 + ch];
        outFinal[(size_t)idx * 2 + ch] = (m0 > 0.5f) ? f : a;
    }
}

// ---------------------------------------------------------------------------
// Host launch
// ---------------------------------------------------------------------------
extern "C" void kernel_launch(void* const* d_in, const int* in_sizes, int n_in,
                              void* d_out, int out_size, void* d_ws, size_t ws_size,
                              hipStream_t stream) {
    (void)in_sizes; (void)n_in; (void)out_size; (void)ws_size;
    const float* x    = (const float*)d_in[0];
    const float* v    = (const float*)d_in[1];
    const float* gum  = (const float*)d_in[2];
    const float* W1   = (const float*)d_in[3];
    const float* b1   = (const float*)d_in[4];
    const float* W2   = (const float*)d_in[5];
    const float* b2   = (const float*)d_in[6];
    const float* cw1  = (const float*)d_in[7];
    const float* cb1  = (const float*)d_in[8];
    const float* cw2  = (const float*)d_in[9];
    const float* cb2  = (const float*)d_in[10];
    const float* cw3  = (const float*)d_in[11];
    const float* cb3  = (const float*)d_in[12];

    char* ws = (char*)d_ws;
    unsigned short* w1th  = (unsigned short*)(ws + 0);
    unsigned short* w1tl  = (unsigned short*)(ws + 1048576);
    float*  scoresBuf     = (float*)(ws + 2097152);
    float*  maskWs        = (float*)(ws + 2359296);
    int*    skip          = (int*)(ws + 2621440);
    double* theta         = (double*)(ws + 2629632);
    float*  feats         = (float*)(ws + 3678720);
    float*  sumMask       = (float*)(ws + 4203008);

    float* out      = (float*)d_out;
    float* finalOut = out;                       // (B,N,2)
    float* maskOut  = out + BB * NN * 2;         // (B,N)
    float* sparsity = out + BB * NN * 3;         // scalar

    k_init<<<257, 256, 0, stream>>>(scoresBuf, sumMask);
    k_prep<<<(HH * DD) / 256, 256, 0, stream>>>(W1, w1th, w1tl);
    k_gemm<<<(BB * NN / 64) * 2, 256, 0, stream>>>(x, w1th, w1tl, b1, W2, scoresBuf);
    k_mask<<<(BB * NN) / 256, 256, 0, stream>>>(scoresBuf, b2, gum, maskOut, maskWs, sumMask);
    k_skip<<<NN / 256, 256, 0, stream>>>(maskWs, skip, sumMask, sparsity);
    k_scan<<<BB, 32, 0, stream>>>(v, skip, theta, feats);
    k_conv<<<(BB * NN) / 256, 256, 0, stream>>>(feats, maskWs, cw1, cb1, cw2, cb2, cw3, cb3, finalOut);
}
